// behavior_model_76046690943718
// MI455X (gfx1250) — matrix operand order repaired and checked by reading, compiled, not run on hardware
//
#include <hip/hip_runtime.h>

// ---------------- types ----------------
typedef __attribute__((ext_vector_type(16))) __bf16 v16bf;
typedef __attribute__((ext_vector_type(8)))  float  v8f;
typedef __attribute__((ext_vector_type(4)))  unsigned int uint4v;
typedef __attribute__((ext_vector_type(4)))  int v4i;

union FragB16 { v16bf v; uint4v u[2]; };

// CDNA5 async global->LDS path (ASYNCcnt-tracked), if the toolchain has it.
#if defined(__gfx1250__) && \
    __has_builtin(__builtin_amdgcn_global_load_async_to_lds_b128) && \
    __has_builtin(__builtin_amdgcn_s_wait_asynccnt)
#define USE_ASYNC_LDS 1
#else
#define USE_ASYNC_LDS 0
#endif

// global_load_async_to_lds_b128 takes (v4i AS1*, v4i AS3*, Ii offset, Ii cpol)
#define GPTR128(p) ((__attribute__((address_space(1))) v4i*)(p))
#define LPTR128(p) ((__attribute__((address_space(3))) v4i*)(p))

// ---------------- problem sizes ----------------
#define BATCH   128
#define NROI    8
#define BNTOT   1024          // BATCH*NROI
#define CIN     1024
#define FSZ     14
#define HW9     9
#define M1      9216          // BNTOT*HW9
#define C1      512
#define C2      256
#define DFEAT   2304          // C2*HW9
#define H1DIM   1024
#define NPID    20
#define OUTC    27

// ---------------- workspace offsets (bytes, 256-aligned) ----------------
#define OFF_ABUF   0ull                      // bf16 [9216][1024]  18,874,368
#define OFF_W1B    18874368ull               // bf16 [512][1024]    1,048,576
#define OFF_OUT1   19922944ull               // bf16 [9216][512]    9,437,184
#define OFF_W2B    29360128ull               // bf16 [256][512]       262,144
#define OFF_FEAT   29622272ull               // f32  [1024][2304]   9,437,184
#define OFF_MEANB  39059456ull               // f32  [20][2304]       184,320
#define OFF_DIFFB  39243776ull               // bf16 [1024][2304]   4,718,592
#define OFF_FC1WB  43962368ull               // bf16 [1024][2304]   4,718,592
#define OFF_H1B    48680960ull               // bf16 [1024][1024]   2,097,152
#define OFF_FC2WB  50778112ull               // bf16 [32][1024]        65,536

// =====================================================================
// ROI-align: one block per (b,n) ROI. Produces bf16 A-matrix rows
// m = bn*9 + hw, k = channel.  6x6 bilinear samples -> 2x2 avg -> 3x3.
// =====================================================================
__global__ __launch_bounds__(256) void roi_align_kernel(
    const float* __restrict__ fmap, const float* __restrict__ boxes,
    __bf16* __restrict__ Abuf)
{
  const int bn = blockIdx.x;           // 0..1023
  const int b  = bn >> 3;
  __shared__ int   soff[36][4];
  __shared__ float swt[36][4];
  const int tid = threadIdx.x & 255;

  if (tid < 36) {
    const float bx1 = boxes[bn * 4 + 0];
    const float by1 = boxes[bn * 4 + 1];
    const float bx2 = boxes[bn * 4 + 2];
    const float by2 = boxes[bn * 4 + 3];
    const float bw = (bx2 - bx1) * (1.0f / 3.0f);
    const float bh = (by2 - by1) * (1.0f / 3.0f);
    const int i = tid / 6, j = tid % 6;
    const float y = by1 + ((float)i + 0.5f) * 0.5f * bh;
    const float x = bx1 + ((float)j + 0.5f) * 0.5f * bw;
    const bool valid = (y > -1.0f) && (y < (float)FSZ) &&
                       (x > -1.0f) && (x < (float)FSZ);
    const float yc = fminf(fmaxf(y, 0.0f), (float)(FSZ - 1));
    const float xc = fminf(fmaxf(x, 0.0f), (float)(FSZ - 1));
    const int y0 = (int)floorf(yc), x0 = (int)floorf(xc);
    const int y1i = min(y0 + 1, FSZ - 1), x1i = min(x0 + 1, FSZ - 1);
    const float ly = yc - (float)y0, lx = xc - (float)x0;
    const float hy = 1.0f - ly, hx = 1.0f - lx;
    soff[tid][0] = y0 * FSZ + x0;   soff[tid][1] = y0 * FSZ + x1i;
    soff[tid][2] = y1i * FSZ + x0;  soff[tid][3] = y1i * FSZ + x1i;
    const float vf = valid ? 1.0f : 0.0f;
    swt[tid][0] = vf * hy * hx;  swt[tid][1] = vf * hy * lx;
    swt[tid][2] = vf * ly * hx;  swt[tid][3] = vf * ly * lx;
  }
  __syncthreads();

  for (int c = tid; c < CIN; c += 256) {
    const float* fp = fmap + (size_t)(b * CIN + c) * (FSZ * FSZ);
    __builtin_prefetch(fp, 0, 3);          // global_prefetch_b8, L2-resident fmap
    float accv[9];
#pragma unroll
    for (int q = 0; q < 9; ++q) accv[q] = 0.0f;
#pragma unroll
    for (int s = 0; s < 36; ++s) {
      const int si = s / 6, sj = s % 6;
      const int hw = (si >> 1) * 3 + (sj >> 1);
      const float v = swt[s][0] * fp[soff[s][0]] + swt[s][1] * fp[soff[s][1]] +
                      swt[s][2] * fp[soff[s][2]] + swt[s][3] * fp[soff[s][3]];
      accv[hw] += v;
    }
#pragma unroll
    for (int hw = 0; hw < 9; ++hw)
      Abuf[(size_t)(bn * HW9 + hw) * CIN + c] = (__bf16)(accv[hw] * 0.25f);
  }
}

// =====================================================================
// Elementwise casts
// =====================================================================
__global__ __launch_bounds__(256) void cast_f32_bf16(
    const float* __restrict__ s, __bf16* __restrict__ d, int n)
{
  const int i = blockIdx.x * 256 + (int)(threadIdx.x & 255);
  if (i < n) d[i] = (__bf16)s[i];
}

__global__ __launch_bounds__(256) void pad_fc2_kernel(
    const float* __restrict__ s, __bf16* __restrict__ d)
{
  const int i = blockIdx.x * 256 + (int)(threadIdx.x & 255);  // 32*1024 total
  const int r = i >> 10, c = i & 1023;
  d[i] = (r < OUTC) ? (__bf16)s[r * 1024 + c] : (__bf16)0.0f;
}

// =====================================================================
// WMMA bf16 GEMM:  out = A(MxK,row-major) * W^T, W is (N,K) row-major,
// so B[k][n] = W[n*K + k].  f32 accumulate via v_wmma_f32_16x16x32_bf16.
// Global->LDS staging uses GLOBAL_LOAD_ASYNC_TO_LDS_B128 (ASYNCcnt) when
// available; double-buffered, one s_wait_asynccnt + barrier per k-step.
//   TM,TN  : 16x16 WMMA tiles per wave (rows, cols)
//   WR,WC  : wave grid inside the 8-wave block
//   EPI    : 0 = bf16 store (ld=Nn)
//            1 = f32 transpose-to-feat store: feat[m/9][n*9 + m%9]
//            2 = bias + relu -> bf16 (ld=Nn)
//            3 = bias, cols<27 only -> f32 (ld=27)
// Requires: M % (WR*TM*16) == 0, Nn % (WC*TN*16) == 0, K % 32 == 0.
// =====================================================================
template <int TM, int TN, int WR, int WC, int EPI>
__global__ __launch_bounds__(256) void gemm_wmma(
    const __bf16* __restrict__ A, const __bf16* __restrict__ W,
    const float* __restrict__ bias, void* __restrict__ outp,
    int M, int Nn, int K)
{
  (void)M;
  constexpr int BM  = WR * TM * 16;
  constexpr int BNt = WC * TN * 16;
  constexpr int LD  = 40;                       // 32 + 8 pad (bf16 units)
  constexpr int ACH = (BM * 4 + 255) / 256;     // 16B chunks per thread (A tile)
  constexpr int WCH = (BNt * 4 + 255) / 256;    // 16B chunks per thread (W tile)

  __shared__ __align__(16) __bf16 sA[2][BM * LD];
  __shared__ __align__(16) __bf16 sW[2][BNt * LD];

  // &255: known-bits -> the idx<BM*4 guards fold away for 128-wide tiles.
  const int tid  = (int)(threadIdx.x & 255);
  const int lane = tid & 31;            // wave32
  const int wid  = tid >> 5;
  const int h    = lane >> 4;           // lane half (WMMA layout)
  const int ln   = lane & 15;
  const int wm0  = (wid / WC) * (TM * 16);
  const int wn0  = (wid % WC) * (TN * 16);
  const int m0   = blockIdx.x * BM;
  const int n0   = blockIdx.y * BNt;

  v8f acc[TM][TN];
  const v8f vzero = {0.f, 0.f, 0.f, 0.f, 0.f, 0.f, 0.f, 0.f};
#pragma unroll
  for (int tm = 0; tm < TM; ++tm)
#pragma unroll
    for (int tn = 0; tn < TN; ++tn) acc[tm][tn] = vzero;

  const int NK = K >> 5;

#if USE_ASYNC_LDS
  auto g_async = [&](int kt, int buf) {
#pragma unroll
    for (int j = 0; j < ACH; ++j) {
      const int idx = tid + j * 256;
      if (idx < BM * 4) {
        const int r = idx >> 2, o = (idx & 3) * 8;
        __builtin_amdgcn_global_load_async_to_lds_b128(
            GPTR128(A + (size_t)(m0 + r) * K + kt * 32 + o),
            LPTR128(&sA[buf][r * LD + o]), 0, 0);
      }
    }
#pragma unroll
    for (int j = 0; j < WCH; ++j) {
      const int idx = tid + j * 256;
      if (idx < BNt * 4) {
        const int r = idx >> 2, o = (idx & 3) * 8;
        __builtin_amdgcn_global_load_async_to_lds_b128(
            GPTR128(W + (size_t)(n0 + r) * K + kt * 32 + o),
            LPTR128(&sW[buf][r * LD + o]), 0, 0);
      }
    }
  };
  g_async(0, 0);
  __builtin_amdgcn_s_wait_asynccnt(0);
  __syncthreads();
#else
  uint4v ra[ACH], rw[WCH];
  auto g_load = [&](int kt) {
#pragma unroll
    for (int j = 0; j < ACH; ++j) {
      const int idx = tid + j * 256;
      if (idx < BM * 4) {
        const int r = idx >> 2, o = (idx & 3) * 8;
        ra[j] = *(const uint4v*)(A + (size_t)(m0 + r) * K + kt * 32 + o);
      }
    }
#pragma unroll
    for (int j = 0; j < WCH; ++j) {
      const int idx = tid + j * 256;
      if (idx < BNt * 4) {
        const int r = idx >> 2, o = (idx & 3) * 8;
        rw[j] = *(const uint4v*)(W + (size_t)(n0 + r) * K + kt * 32 + o);
      }
    }
  };
  auto s_store = [&](int buf) {
#pragma unroll
    for (int j = 0; j < ACH; ++j) {
      const int idx = tid + j * 256;
      if (idx < BM * 4) {
        const int r = idx >> 2, o = (idx & 3) * 8;
        *(uint4v*)&sA[buf][r * LD + o] = ra[j];
      }
    }
#pragma unroll
    for (int j = 0; j < WCH; ++j) {
      const int idx = tid + j * 256;
      if (idx < BNt * 4) {
        const int r = idx >> 2, o = (idx & 3) * 8;
        *(uint4v*)&sW[buf][r * LD + o] = rw[j];
      }
    }
  };
  g_load(0);
  s_store(0);
  __syncthreads();
#endif

  for (int kt = 0; kt < NK; ++kt) {
    const int cur = kt & 1;
#if USE_ASYNC_LDS
    if (kt + 1 < NK) g_async(kt + 1, cur ^ 1);  // overlap fetch with WMMA
#else
    if (kt + 1 < NK) g_load(kt + 1);
#endif

    FragB16 fa[TM], fb[TN];
#pragma unroll
    for (int tm = 0; tm < TM; ++tm) {
      // ISA 16-bit A 16x32 layout: half0 -> K 0-7 & 16-23, half1 -> K 8-15 & 24-31
      const __bf16* p = &sA[cur][(wm0 + tm * 16 + ln) * LD];
      fa[tm].u[0] = *(const uint4v*)(p + h * 8);
      fa[tm].u[1] = *(const uint4v*)(p + 16 + h * 8);
    }
#pragma unroll
    for (int tn = 0; tn < TN; ++tn) {
      // B 32x16 layout matching A: half0 -> K 0-7 & 16-23, half1 -> K 8-15 & 24-31
      const __bf16* p = &sW[cur][(wn0 + tn * 16 + ln) * LD + h * 8];
      fb[tn].u[0] = *(const uint4v*)(p);
      fb[tn].u[1] = *(const uint4v*)(p + 16);
    }
#pragma unroll
    for (int tm = 0; tm < TM; ++tm)
#pragma unroll
      for (int tn = 0; tn < TN; ++tn)
        acc[tm][tn] = __builtin_amdgcn_wmma_f32_16x16x32_bf16(
            false, fa[tm].v, false, fb[tn].v, (short)0, acc[tm][tn],
            false, false);

#if USE_ASYNC_LDS
    if (kt + 1 < NK) {
      __builtin_amdgcn_s_wait_asynccnt(0);   // tile kt+1 resident in LDS
      __syncthreads();
    }
#else
    if (kt + 1 < NK) {
      __syncthreads();
      s_store(cur ^ 1);
      __syncthreads();
    }
#endif
  }

  // ---------------- epilogue ----------------
#pragma unroll
  for (int tm = 0; tm < TM; ++tm) {
#pragma unroll
    for (int tn = 0; tn < TN; ++tn) {
#pragma unroll
      for (int v = 0; v < 8; ++v) {
        const int m = m0 + wm0 + tm * 16 + h * 8 + v;  // C/D: VGPR v -> row
        const int n = n0 + wn0 + tn * 16 + ln;         // lane -> col
        const float val = acc[tm][tn][v];
        if constexpr (EPI == 0) {
          ((__bf16*)outp)[(size_t)m * Nn + n] = (__bf16)val;
        } else if constexpr (EPI == 1) {
          const int bn = m / 9, hw = m - bn * 9;
          ((float*)outp)[(size_t)bn * DFEAT + n * 9 + hw] = val;
        } else if constexpr (EPI == 2) {
          float x = val + bias[n];
          x = x > 0.0f ? x : 0.0f;
          ((__bf16*)outp)[(size_t)m * Nn + n] = (__bf16)x;
        } else {
          if (n < OUTC) ((float*)outp)[(size_t)m * OUTC + n] = val + bias[n];
        }
      }
    }
  }
}

// =====================================================================
// mean over batch of pid-scattered features:
// mean_b[p][d] = (1/128) * sum_{(b,n): pids[b,n]==p} feat[b*8+n][d]
// (pids are distinct within a batch -> at most 128 members per p)
// =====================================================================
__global__ __launch_bounds__(256) void mean_kernel(
    const float* __restrict__ feat, const int* __restrict__ pids,
    float* __restrict__ meanb)
{
  __shared__ int list[BATCH];
  __shared__ int cnt;
  const int p = blockIdx.x, tid = (int)(threadIdx.x & 255);
  if (tid == 0) cnt = 0;
  __syncthreads();
  for (int e = tid; e < BNTOT; e += 256)
    if (pids[e] == p) {
      const int i = atomicAdd(&cnt, 1);
      if (i < BATCH) list[i] = e;
    }
  __syncthreads();
  int nloc = cnt;
  if (nloc > BATCH) nloc = BATCH;
  for (int d = tid; d < DFEAT; d += 256) {
    float s = 0.0f;
    for (int i = 0; i < nloc; ++i) s += feat[(size_t)list[i] * DFEAT + d];
    meanb[p * DFEAT + d] = s * (1.0f / (float)BATCH);
  }
}

__global__ __launch_bounds__(256) void diff_kernel(
    const float* __restrict__ feat, const float* __restrict__ meanb,
    const int* __restrict__ pids, __bf16* __restrict__ diffb)
{
  const int bn = blockIdx.x;
  const int d  = blockIdx.y * 256 + (int)(threadIdx.x & 255);  // DFEAT = 9*256
  const int p  = pids[bn];
  diffb[(size_t)bn * DFEAT + d] =
      (__bf16)(meanb[p * DFEAT + d] - feat[(size_t)bn * DFEAT + d]);
}

// =====================================================================
extern "C" void kernel_launch(void* const* d_in, const int* in_sizes, int n_in,
                              void* d_out, int out_size, void* d_ws,
                              size_t ws_size, hipStream_t stream)
{
  (void)in_sizes; (void)n_in; (void)out_size; (void)ws_size;
  const float* fmap  = (const float*)d_in[0];
  const float* boxes = (const float*)d_in[1];
  const int*   pids  = (const int*)d_in[2];
  const float* w1    = (const float*)d_in[3];
  const float* w2    = (const float*)d_in[4];
  const float* fc1w  = (const float*)d_in[5];
  const float* fc1b  = (const float*)d_in[6];
  const float* fc2w  = (const float*)d_in[7];
  const float* fc2b  = (const float*)d_in[8];

  char* ws = (char*)d_ws;
  __bf16* Abuf  = (__bf16*)(ws + OFF_ABUF);
  __bf16* w1b   = (__bf16*)(ws + OFF_W1B);
  __bf16* out1  = (__bf16*)(ws + OFF_OUT1);
  __bf16* w2b   = (__bf16*)(ws + OFF_W2B);
  float*  featf = (float*)(ws + OFF_FEAT);
  float*  meanb = (float*)(ws + OFF_MEANB);
  __bf16* diffb = (__bf16*)(ws + OFF_DIFFB);
  __bf16* fc1wb = (__bf16*)(ws + OFF_FC1WB);
  __bf16* h1b   = (__bf16*)(ws + OFF_H1B);
  __bf16* fc2wb = (__bf16*)(ws + OFF_FC2WB);

  // Stage 0: ROI-align -> bf16 A matrix; weight casts (all independent).
  roi_align_kernel<<<BNTOT, 256, 0, stream>>>(fmap, boxes, Abuf);
  cast_f32_bf16<<<(C1 * CIN) / 256, 256, 0, stream>>>(w1, w1b, C1 * CIN);
  cast_f32_bf16<<<(C2 * C1) / 256, 256, 0, stream>>>(w2, w2b, C2 * C1);
  cast_f32_bf16<<<(H1DIM * DFEAT) / 256, 256, 0, stream>>>(fc1w, fc1wb,
                                                           H1DIM * DFEAT);
  pad_fc2_kernel<<<(32 * 1024) / 256, 256, 0, stream>>>(fc2w, fc2wb);

  // GEMM1: (9216x1024)x(1024->512) -> out1 bf16
  gemm_wmma<2, 4, 4, 2, 0><<<dim3(M1 / 128, C1 / 128), 256, 0, stream>>>(
      Abuf, w1b, nullptr, out1, M1, C1, CIN);

  // GEMM2: (9216x512)x(512->256) -> feat f32 (channel-major flatten)
  gemm_wmma<2, 4, 4, 2, 1><<<dim3(M1 / 128, C2 / 128), 256, 0, stream>>>(
      out1, w2b, nullptr, featf, M1, C2, C1);

  // pid mean + diff -> bf16
  mean_kernel<<<NPID, 256, 0, stream>>>(featf, pids, meanb);
  diff_kernel<<<dim3(BNTOT, DFEAT / 256), 256, 0, stream>>>(featf, meanb, pids,
                                                            diffb);

  // FC1: (1024x2304)x(2304->1024) + bias + relu -> h1 bf16
  gemm_wmma<2, 4, 4, 2, 2><<<dim3(BNTOT / 128, H1DIM / 128), 256, 0, stream>>>(
      diffb, fc1wb, fc1b, h1b, BNTOT, H1DIM, DFEAT);

  // FC2: (1024x1024)x(1024->32, 27 valid) + bias -> d_out f32
  gemm_wmma<1, 2, 8, 1, 3><<<dim3(BNTOT / 128, 1), 256, 0, stream>>>(
      h1b, fc2wb, fc2b, d_out, BNTOT, 32, CIN);
}
